// LinAttn_21208548507758
// MI455X (gfx1250) — compile-verified, matrix operand order checked by reading, not run on hardware
//
#include <hip/hip_runtime.h>
#include <hip/hip_bf16.h>

// ---------------------------------------------------------------------------
// Taylor-feature causal linear attention for MI455X (gfx1250, wave32, WMMA).
// Factorized state: S0=Σv, S1=Σ k vᵀ/4, S2=Σ vec(kkᵀ) vᵀ/32 (256x16),
//                   Z1=Σ k/4, Z2=Σ kkᵀ/32, z0=count.
// phi(q)·phi(k) = 1 + (q·k)/4 + (q·k)²/32, so the 273-dim feature map is
// never materialized. Segment-parallel 3-phase scan (states are additive).
// ---------------------------------------------------------------------------

typedef __attribute__((ext_vector_type(16))) _Float16 v16h;
typedef __attribute__((ext_vector_type(8)))  float    v8f;

#define B_SZ    4
#define L_SZ    2048
#define DMODEL  32
#define NH      2
#define BH      (B_SZ*NH)        // 8 sequences
#define NSEG    16
#define SEG_LEN (L_SZ/NSEG)      // 128
#define NCHUNK  (SEG_LEN/16)     // 8 chunks of 16 per segment

// flattened state layout (floats)
#define S0_OFF  0
#define Z1_OFF  16
#define S1_OFF  32
#define Z2_OFF  288
#define S2_OFF  544
#define SSZ     4640

__device__ __forceinline__ int a_k0(int v, int half) {
  return ((v < 4) ? 2 * v : 16 + 2 * (v - 4)) + 8 * half;
}

__device__ __forceinline__ v8f v8zero() {
  v8f c;
#pragma unroll
  for (int i = 0; i < 8; ++i) c[i] = 0.0f;
  return c;
}

// A fragment, row-major 16x16 LDS source, K dim = cols (pad K=16..31 with 0)
__device__ __forceinline__ v16h fragA16(const float* m, int lane, float scale) {
  int mr = lane & 15, half = lane >> 4;
  v16h a;
#pragma unroll
  for (int v = 0; v < 8; ++v) {
    int k0 = a_k0(v, half);
    a[2 * v]     = (k0     < 16) ? (_Float16)(scale * m[mr * 16 + k0])     : (_Float16)0.0f;
    a[2 * v + 1] = (k0 + 1 < 16) ? (_Float16)(scale * m[mr * 16 + k0 + 1]) : (_Float16)0.0f;
  }
  return a;
}

// A fragment = transpose of 16x16 LDS source: A[m][k] = src[k*16+m]
__device__ __forceinline__ v16h fragAT16(const float* m, int lane, float scale) {
  int mr = lane & 15, half = lane >> 4;
  v16h a;
#pragma unroll
  for (int v = 0; v < 8; ++v) {
    int k0 = a_k0(v, half);
    a[2 * v]     = (k0     < 16) ? (_Float16)(scale * m[k0 * 16 + mr])       : (_Float16)0.0f;
    a[2 * v + 1] = (k0 + 1 < 16) ? (_Float16)(scale * m[(k0 + 1) * 16 + mr]) : (_Float16)0.0f;
  }
  return a;
}

// A fragment block of Q2 = vec(q qᵀ): A[m][k] = Q[m][(32b+k)/16] * Q[m][(32b+k)%16]
__device__ __forceinline__ v16h fragA_q2(const float* Q, int lane, int blk) {
  int mr = lane & 15, half = lane >> 4;
  v16h a;
#pragma unroll
  for (int v = 0; v < 8; ++v) {
    int k0 = a_k0(v, half);
    int f0 = 32 * blk + k0, f1 = f0 + 1;
    a[2 * v]     = (_Float16)(Q[mr * 16 + (f0 >> 4)] * Q[mr * 16 + (f0 & 15)]);
    a[2 * v + 1] = (_Float16)(Q[mr * 16 + (f1 >> 4)] * Q[mr * 16 + (f1 & 15)]);
  }
  return a;
}

// A fragment block j of K2ᵀ: A[m][k] = K[k][j]*K[k][m], k = time < 16
__device__ __forceinline__ v16h fragA_k2t(const float* K, int lane, int j, float scale) {
  int mr = lane & 15, half = lane >> 4;
  v16h a;
#pragma unroll
  for (int v = 0; v < 8; ++v) {
    int k0 = a_k0(v, half);
    a[2 * v]     = (k0     < 16) ? (_Float16)(scale * K[k0 * 16 + j] * K[k0 * 16 + mr])             : (_Float16)0.0f;
    a[2 * v + 1] = (k0 + 1 < 16) ? (_Float16)(scale * K[(k0 + 1) * 16 + j] * K[(k0 + 1) * 16 + mr]) : (_Float16)0.0f;
  }
  return a;
}

// B fragment, rows = K (kvalid rows valid, stride 16): B[k][n] = src[k*16+n]
__device__ __forceinline__ v16h fragB16(const float* src, int lane, int kvalid, float scale) {
  int n = lane & 15, half = lane >> 4;
  v16h b;
#pragma unroll
  for (int v = 0; v < 8; ++v) {
    int k0 = a_k0(v, half);
    b[2 * v]     = (k0     < kvalid) ? (_Float16)(scale * src[k0 * 16 + n])       : (_Float16)0.0f;
    b[2 * v + 1] = (k0 + 1 < kvalid) ? (_Float16)(scale * src[(k0 + 1) * 16 + n]) : (_Float16)0.0f;
  }
  return b;
}

// B fragment = transpose: B[k][n] = src[n*16+k]  (for Kᵀ on the B side)
__device__ __forceinline__ v16h fragBT16(const float* src, int lane, int kvalid) {
  int n = lane & 15, half = lane >> 4;
  v16h b;
#pragma unroll
  for (int v = 0; v < 8; ++v) {
    int k0 = a_k0(v, half);
    b[2 * v]     = (k0     < kvalid) ? (_Float16)(src[n * 16 + k0])     : (_Float16)0.0f;
    b[2 * v + 1] = (k0 + 1 < kvalid) ? (_Float16)(src[n * 16 + k0 + 1]) : (_Float16)0.0f;
  }
  return b;
}

__device__ __forceinline__ v8f wmma16(v16h a, v16h b, v8f c) {
  return __builtin_amdgcn_wmma_f32_16x16x32_f16(false, a, false, b, (short)0, c, false, false);
}

__device__ __forceinline__ v8f loadC(const float* src, int lane) {
  int n = lane & 15, half = lane >> 4;
  v8f c;
#pragma unroll
  for (int r = 0; r < 8; ++r) c[r] = src[(r + 8 * half) * 16 + n];
  return c;
}
__device__ __forceinline__ void storeC(float* dst, v8f c, int lane) {
  int n = lane & 15, half = lane >> 4;
#pragma unroll
  for (int r = 0; r < 8; ++r) dst[(r + 8 * half) * 16 + n] = c[r];
}

// ---------------------------------------------------------------------------
// Kernel 1: QKV projections. q/k/v layout: [(bh)*L + t]*16 + f, bh = b*NH + h
// ---------------------------------------------------------------------------
__global__ void la_qkv_proj(const float* __restrict__ x, const float* __restrict__ Wq,
                            const float* __restrict__ Wk, const float* __restrict__ Wv,
                            float* __restrict__ q, float* __restrict__ k, float* __restrict__ v) {
  int idx = blockIdx.x * blockDim.x + threadIdx.x;
  if (idx >= B_SZ * L_SZ * 96) return;
  int tok = idx / 96, o = idx % 96;
  int b = tok / L_SZ, l = tok % L_SZ;
  const float* xr = x + (size_t)tok * DMODEL;
  const float* w;
  float* dst;
  int oo;
  if (o < 32)       { oo = o;      w = Wq + oo * DMODEL; dst = q; }
  else if (o < 64)  { oo = o - 32; w = Wk + oo * DMODEL; dst = k; }
  else              { oo = o - 64; w = Wv + oo * DMODEL; dst = v; }
  float acc = 0.0f;
#pragma unroll
  for (int c = 0; c < DMODEL; ++c) acc += xr[c] * w[c];
  int h = oo >> 4, f = oo & 15;
  dst[((size_t)(b * NH + h) * L_SZ + l) * 16 + f] = acc;
}

// ---------------------------------------------------------------------------
// Kernel 2 (phase A): per-(bh,segment) raw state sums. 1 wave / block.
// State accumulators live in C-layout registers (S2 = 16 v8f tiles).
// ---------------------------------------------------------------------------
__global__ void __launch_bounds__(32)
la_seg_state(const float* __restrict__ gk, const float* __restrict__ gv,
             float* __restrict__ raw) {
  int lane = threadIdx.x;
  int bh = blockIdx.x / NSEG, seg = blockIdx.x % NSEG;
  __shared__ float Kc[256], Vc[256], red[32];

  v8f s1acc = v8zero(), z2acc = v8zero();
  v8f s2acc[16];
#pragma unroll
  for (int j = 0; j < 16; ++j) s2acc[j] = v8zero();
  float s0part = 0.0f, z1part = 0.0f;

  int n = lane & 15, half = lane >> 4;
  for (int cc = 0; cc < NCHUNK; ++cc) {
    int t0 = seg * SEG_LEN + cc * 16;
    size_t gbase = ((size_t)bh * L_SZ + t0) * 16;
    for (int i = lane; i < 256; i += 32) { Kc[i] = gk[gbase + i]; Vc[i] = gv[gbase + i]; }
    __syncthreads();

    v16h aKt = fragAT16(Kc, lane, 0.25f);        // (K/4)ᵀ
    v16h bV  = fragB16(Vc, lane, 16, 1.0f);
    s1acc = wmma16(aKt, bV, s1acc);              // S1 += KᵀV/4
    z2acc = wmma16(aKt, fragB16(Kc, lane, 16, 0.125f), z2acc);  // Z2 += KᵀK/32
#pragma unroll
    for (int j = 0; j < 16; ++j)
      s2acc[j] = wmma16(fragA_k2t(Kc, lane, j, 1.0f / 32.0f), bV, s2acc[j]);

#pragma unroll
    for (int t = 0; t < 8; ++t) {
      s0part += Vc[(8 * half + t) * 16 + n];
      z1part += Kc[(8 * half + t) * 16 + n];
    }
    __syncthreads();
  }

  size_t sb = (size_t)(bh * NSEG + seg) * SSZ;
#pragma unroll
  for (int r = 0; r < 8; ++r) {
    raw[sb + S1_OFF + (r + 8 * half) * 16 + n] = s1acc[r];
    raw[sb + Z2_OFF + (r + 8 * half) * 16 + n] = z2acc[r];
  }
#pragma unroll
  for (int j = 0; j < 16; ++j)
#pragma unroll
    for (int r = 0; r < 8; ++r)
      raw[sb + S2_OFF + (16 * j + r + 8 * half) * 16 + n] = s2acc[j][r];

  red[lane] = s0part; __syncthreads();
  if (lane < 16) raw[sb + S0_OFF + lane] = red[lane] + red[lane + 16];
  __syncthreads();
  red[lane] = z1part; __syncthreads();
  if (lane < 16) raw[sb + Z1_OFF + lane] = 0.25f * (red[lane] + red[lane + 16]);
}

// ---------------------------------------------------------------------------
// Kernel 3 (phase B): exclusive prefix over the 16 segments of each sequence.
// ---------------------------------------------------------------------------
__global__ void la_seg_scan(const float* __restrict__ raw, float* __restrict__ pref) {
  int bh = blockIdx.x;
  for (int e = threadIdx.x; e < SSZ; e += blockDim.x) {
    float run = 0.0f;
    for (int s = 0; s < NSEG; ++s) {
      size_t idx = (size_t)(bh * NSEG + s) * SSZ + e;
      float t = raw[idx];
      pref[idx] = run;
      run += t;
    }
  }
}

// ---------------------------------------------------------------------------
// Kernel 4 (phase C): full chunked linear attention inside each segment.
// ---------------------------------------------------------------------------
__global__ void __launch_bounds__(32)
la_attn_main(const float* __restrict__ gq, const float* __restrict__ gk,
             const float* __restrict__ gv, const float* __restrict__ pref,
             float* __restrict__ gy) {
  int lane = threadIdx.x;
  int bh = blockIdx.x / NSEG, seg = blockIdx.x % NSEG;
  int n = lane & 15, half = lane >> 4;

  __shared__ float ST[SSZ];              // running state (S0,Z1,S1,Z2,S2)
  __shared__ float Qc[256], Kc[256], Vc[256], Mm[256], Wp[256];
  __shared__ float dI[16], Den[16];

  size_t sb = (size_t)(bh * NSEG + seg) * SSZ;
  for (int i = lane; i < SSZ; i += 32) ST[i] = pref[sb + i];
  __syncthreads();

  for (int cc = 0; cc < NCHUNK; ++cc) {
    int t0 = seg * SEG_LEN + cc * 16;
    float prevcnt = (float)(seg * SEG_LEN + cc * 16);
    size_t gbase = ((size_t)bh * L_SZ + t0) * 16;
    for (int i = lane; i < 256; i += 32) {
      Qc[i] = gq[gbase + i]; Kc[i] = gk[gbase + i]; Vc[i] = gv[gbase + i];
    }
    __syncthreads();

    // --- s = Q Kᵀ ------------------------------------------------------
    v16h aQ = fragA16(Qc, lane, 1.0f);
    v8f  s  = wmma16(aQ, fragBT16(Kc, lane, 16), v8zero());

    // --- polynomial kernel + causal mask ------------------------------
#pragma unroll
    for (int r = 0; r < 8; ++r) {
      int row = r + 8 * half;
      float sv = s[r];
      float av = 1.0f + 0.25f * sv + sv * sv * (1.0f / 32.0f);
      Mm[row * 16 + n] = (n <= row) ? av : 0.0f;
    }
    __syncthreads();
    if (lane < 16) {
      float di = 0.0f;
#pragma unroll
      for (int j = 0; j < 16; ++j) di += Mm[lane * 16 + j];
      dI[lane] = di;
    }
    __syncthreads();

    // --- numerator: S0 + M·V + Q·S1 + Q2·S2 ---------------------------
    v8f cnum;
#pragma unroll
    for (int r = 0; r < 8; ++r) cnum[r] = ST[S0_OFF + n];
    v16h bV = fragB16(Vc, lane, 16, 1.0f);
    cnum = wmma16(fragA16(Mm, lane, 1.0f), bV, cnum);
    cnum = wmma16(aQ, fragB16(ST + S1_OFF, lane, 16, 1.0f), cnum);
    for (int i = 0; i < 8; ++i)
      cnum = wmma16(fragA_q2(Qc, lane, i), fragB16(ST + S2_OFF + 32 * i * 16, lane, 32, 1.0f), cnum);

    // --- denominator: eps + z0 + q·Z1 + qᵀZ2q + intra -----------------
    v8f w = wmma16(aQ, fragB16(ST + Z2_OFF, lane, 16, 1.0f), v8zero());
#pragma unroll
    for (int r = 0; r < 8; ++r) {
      int row = r + 8 * half;
      Wp[row * 16 + n] = w[r] * Qc[row * 16 + n];
    }
    __syncthreads();
    if (lane < 16) {
      float d = 1e-12f + prevcnt + dI[lane];
#pragma unroll
      for (int j = 0; j < 16; ++j)
        d += Qc[lane * 16 + j] * ST[Z1_OFF + j] + Wp[lane * 16 + j];
      Den[lane] = d;
    }
    __syncthreads();

    // --- output chunk --------------------------------------------------
#pragma unroll
    for (int r = 0; r < 8; ++r) {
      int row = r + 8 * half;
      gy[gbase + row * 16 + n] = cnum[r] / Den[row];
    }

    // --- state update (after all query-side reads) --------------------
    v16h aKt = fragAT16(Kc, lane, 0.25f);
    v8f c1 = loadC(ST + S1_OFF, lane);
    c1 = wmma16(aKt, bV, c1);
    storeC(ST + S1_OFF, c1, lane);
    v8f c2 = loadC(ST + Z2_OFF, lane);
    c2 = wmma16(aKt, fragB16(Kc, lane, 16, 0.125f), c2);
    storeC(ST + Z2_OFF, c2, lane);
    for (int j = 0; j < 16; ++j) {
      v8f c3 = loadC(ST + S2_OFF + j * 256, lane);
      c3 = wmma16(fragA_k2t(Kc, lane, j, 1.0f / 32.0f), bV, c3);
      storeC(ST + S2_OFF + j * 256, c3, lane);
    }
    if (lane < 16) {
      float sv = 0.0f, kv = 0.0f;
#pragma unroll
      for (int t = 0; t < 16; ++t) { sv += Vc[t * 16 + lane]; kv += Kc[t * 16 + lane]; }
      ST[S0_OFF + lane] += sv;
      ST[Z1_OFF + lane] += 0.25f * kv;
    }
    __syncthreads();
  }
}

// ---------------------------------------------------------------------------
// Kernel 5: output projection with Wo.
// ---------------------------------------------------------------------------
__global__ void la_out_proj(const float* __restrict__ gy, const float* __restrict__ Wo,
                            float* __restrict__ out) {
  int idx = blockIdx.x * blockDim.x + threadIdx.x;
  if (idx >= B_SZ * L_SZ * DMODEL) return;
  int tok = idx / DMODEL, m = idx % DMODEL;
  int b = tok / L_SZ, l = tok % L_SZ;
  float acc = 0.0f;
#pragma unroll
  for (int j = 0; j < DMODEL; ++j) {
    int h = j >> 4, d = j & 15;
    acc += gy[((size_t)(b * NH + h) * L_SZ + l) * 16 + d] * Wo[m * DMODEL + j];
  }
  out[(size_t)tok * DMODEL + m] = acc;
}

// ---------------------------------------------------------------------------
extern "C" void kernel_launch(void* const* d_in, const int* in_sizes, int n_in,
                              void* d_out, int out_size, void* d_ws, size_t ws_size,
                              hipStream_t stream) {
  (void)in_sizes; (void)n_in; (void)out_size; (void)ws_size;
  const float* x  = (const float*)d_in[0];
  const float* Wq = (const float*)d_in[1];
  const float* Wk = (const float*)d_in[2];
  const float* Wv = (const float*)d_in[3];
  const float* Wo = (const float*)d_in[4];
  float* out = (float*)d_out;

  float* ws = (float*)d_ws;
  const size_t QKV = (size_t)BH * L_SZ * 16;   // 262144 floats each
  float* q    = ws;
  float* k    = q + QKV;
  float* v    = k + QKV;
  float* y    = v + QKV;
  float* raw  = y + QKV;                       // 8*16*4640 floats
  float* pref = raw + (size_t)BH * NSEG * SSZ;

  la_qkv_proj<<<(B_SZ * L_SZ * 96 + 255) / 256, 256, 0, stream>>>(x, Wq, Wk, Wv, q, k, v);
  la_seg_state<<<BH * NSEG, 32, 0, stream>>>(k, v, raw);
  la_seg_scan<<<BH, 256, 0, stream>>>(raw, pref);
  la_attn_main<<<BH * NSEG, 32, 0, stream>>>(q, k, v, pref, y);
  la_out_proj<<<(B_SZ * L_SZ * DMODEL + 255) / 256, 256, 0, stream>>>(y, Wo, out);
}
